// EnhancedEEGS4Model_16509854286057
// MI455X (gfx1250) — matrix operand order repaired and checked by reading, not compiled, not run on hardware
//
#include <hip/hip_runtime.h>
#include <hip/hip_bf16.h>
#include <math.h>
#include <stdint.h>

// ---------------------------------------------------------------------------
// EnhancedEEGS4Model on MI455X (gfx1250, wave32, WMMA)
//   - bf16 WMMA GEMMs for input projection and K/V projection
//   - f32 WMMA (16x16x4) for the sequential S4 state recurrence (LDS-resident)
//   - async-to-LDS (ASYNCcnt) staging for the causal depthwise conv
//   - all intermediates sized to stay resident in the 192MB L2
// ---------------------------------------------------------------------------

typedef __attribute__((ext_vector_type(16))) __bf16 v16bf;
typedef __attribute__((ext_vector_type(8)))  float  v8f;
typedef __attribute__((ext_vector_type(2)))  float  v2f;

#define B_   32
#define C_   129
#define T_   1000
#define D_   256
#define NS_  64
#define NL_  4
#define H_   8
#define HD_  32
#define S_   1001
#define TP_  1024
#define KP1_ 160          // 129 padded to multiple of 32
#define LN_EPS_ 1e-5f

// workspace layout (float element offsets)
#define F_H    0u          // h buffer (B*T, D) f32            : 8,192,000
#define F_X    8388608u    // normalized transposed x (B*D,TP) : 8,388,608
#define F_Y    16777216u   // conv output (B*D,TP)             : 8,388,608
#define F_KV   F_X         // K|V projection out (32032,512) — reuses X+Y after layers
#define F_K1   25165824u   // s4 kernels (NL, D, TP)           : 1,048,576
#define F_Q    26214400u   // q vector (256)
#define F_POOL 26214912u   // pooled (B, D)
#define F_END  26223104u   // bf16 staging region starts here (bytes = 4*F_END)

// ---------------------------------------------------------------------------
// CDNA5 async copy: global -> LDS, 16B per active lane, tracked by ASYNCcnt
// ---------------------------------------------------------------------------
__device__ __forceinline__ void async_load_f4(const float* g, float* l) {
  unsigned lds = (unsigned)(uintptr_t)l;   // flat-LDS aperture keeps offset in addr[31:0]
  asm volatile("global_load_async_to_lds_b128 %0, %1, off"
               :: "v"(lds), "v"(g) : "memory");
}
__device__ __forceinline__ void async_wait0() {
  asm volatile("s_wait_asynccnt 0x0" ::: "memory");
}

// ---------------------------------------------------------------------------
// block reductions (blockDim.x == 256)
// ---------------------------------------------------------------------------
__device__ __forceinline__ float block_sum_256(float v, float* red) {
  int tid = threadIdx.x;
  red[tid] = v; __syncthreads();
  for (int s = 128; s > 0; s >>= 1) { if (tid < s) red[tid] += red[tid + s]; __syncthreads(); }
  float r = red[0]; __syncthreads();
  return r;
}
__device__ __forceinline__ float block_max_256(float v, float* red) {
  int tid = threadIdx.x;
  red[tid] = v; __syncthreads();
  for (int s = 128; s > 0; s >>= 1) { if (tid < s) red[tid] = fmaxf(red[tid], red[tid + s]); __syncthreads(); }
  float r = red[0]; __syncthreads();
  return r;
}

// ---------------------------------------------------------------------------
// bf16 packing kernels
// ---------------------------------------------------------------------------
__global__ void pack_x_kernel(const float* __restrict__ x, __bf16* __restrict__ A) {
  unsigned idx = blockIdx.x * 256u + threadIdx.x;          // < 32000*160
  unsigned row = idx / KP1_, c = idx % KP1_;
  unsigned b = row / T_, t = row % T_;
  float v = (c < C_) ? x[(b * C_ + c) * T_ + t] : 0.0f;    // x is (B,C,T); A row = x[b,:,t]
  A[idx] = (__bf16)v;
}

__global__ void pack_w_kernel(const float* __restrict__ w_in, __bf16* __restrict__ Bt) {
  unsigned idx = blockIdx.x * 256u + threadIdx.x;          // < 256*160, Bt is N x Kp (B^T)
  unsigned d = idx / KP1_, c = idx % KP1_;
  float v = (c < C_) ? w_in[d * C_ + c] : 0.0f;
  Bt[idx] = (__bf16)v;
}

__global__ void pack_kv_kernel(const float* __restrict__ h, const float* __restrict__ cls,
                               __bf16* __restrict__ A) {
  unsigned idx = blockIdx.x * 256u + threadIdx.x;          // < 32032*256
  unsigned r = idx >> 8, k = idx & 255u;
  unsigned b = r / S_, s = r % S_;
  float v = (s == 0) ? cls[k] : h[(b * T_ + (s - 1)) * D_ + k];
  A[idx] = (__bf16)v;
}

__global__ void pack_wkv_kernel(const float* __restrict__ mha_in_w, __bf16* __restrict__ Bt) {
  unsigned idx = blockIdx.x * 256u + threadIdx.x;          // < 512*256  (rows: 0..255 -> wk, 256..511 -> wv)
  unsigned n = idx >> 8, k = idx & 255u;
  Bt[idx] = (__bf16)mha_in_w[(256u + n) * D_ + k];
}

// ---------------------------------------------------------------------------
// generic bf16 WMMA GEMM:  C[M,Nw] = A[M,Kp] * Bt[N,Kp]^T + bias[N]
// one 16x16 output tile per wave, 8 waves per block
// ---------------------------------------------------------------------------
__global__ void gemm_bf16_kernel(const __bf16* __restrict__ A, const __bf16* __restrict__ Bt,
                                 const float* __restrict__ bias, float* __restrict__ Cmat,
                                 int Mtiles, int Ntiles, int Kp, int Nw) {
  (void)Mtiles;
  int lane = threadIdx.x & 31;
  int wv   = threadIdx.x >> 5;
  int tile = blockIdx.x * 8 + wv;
  int mt = tile / Ntiles, nt = tile % Ntiles;
  int m0 = mt * 16, n0 = nt * 16;
  int ln15 = lane & 15, lh = lane >> 4;

  // A fragment: lane holds row m0+(lane&15); K halves {lh*8..+7} and {16+lh*8..+7}
  const __bf16* arow = A + (size_t)(m0 + ln15) * Kp + lh * 8;
  // B fragment: lane holds col n0+(lane&15); K halves {lh*8..+7} and {16+lh*8..+7} (Bt stored transposed)
  const __bf16* brow = Bt + (size_t)(n0 + ln15) * Kp + lh * 8;

  v8f acc = {0.f, 0.f, 0.f, 0.f, 0.f, 0.f, 0.f, 0.f};
  for (int kc = 0; kc < Kp; kc += 32) {
    union { v16bf v; uint4 q[2]; } ua, ub;
    ua.q[0] = *(const uint4*)(arow + kc);        // K kc+lh*8 .. +7
    ua.q[1] = *(const uint4*)(arow + kc + 16);   // K kc+16+lh*8 .. +7
    ub.q[0] = *(const uint4*)(brow + kc);        // K kc+lh*8 .. +7
    ub.q[1] = *(const uint4*)(brow + kc + 16);   // K kc+16+lh*8 .. +7
    __builtin_prefetch(arow + kc + 128, 0, 1);   // global_prefetch_b8
    acc = __builtin_amdgcn_wmma_f32_16x16x32_bf16(false, ua.v, false, ub.v,
                                                  (short)0, acc, false, false);
  }
  int col = n0 + ln15;
  float bc = bias[col];
  for (int v = 0; v < 8; ++v) {
    int row = m0 + v + lh * 8;                   // C/D layout: VGPR v -> rows v and v+8
    Cmat[(size_t)row * Nw + col] = acc[v] + bc;
  }
}

// ---------------------------------------------------------------------------
// LayerNorm + positional encoding (in place, one (b,t) row per block)
// ---------------------------------------------------------------------------
__global__ void ln_pos_kernel(float* __restrict__ h, const float* __restrict__ g,
                              const float* __restrict__ bta) {
  __shared__ float red[256];
  unsigned row = blockIdx.x;
  unsigned t = row % T_;
  int d = threadIdx.x;
  float v = h[(size_t)row * D_ + d];
  float mean = block_sum_256(v, red) * (1.0f / D_);
  float dv = v - mean;
  float var = block_sum_256(dv * dv, red) * (1.0f / D_);
  float y = dv * rsqrtf(var + LN_EPS_) * g[d] + bta[d];
  float ex = -logf(10000.0f) * (float)(d & ~1) / (float)D_;
  float ang = (float)t * expf(ex);
  y += (d & 1) ? cosf(ang) : sinf(ang);
  h[(size_t)row * D_ + d] = y;
}

// ---------------------------------------------------------------------------
// S4 kernel generation: 64x64 solve (Gauss-Jordan) + 1000-step scan with
// f32 WMMA 16x16x4 on LDS-resident state. One block per layer.
// dynamic LDS: AdT[64*66] + xc[256*66]  = 84,480 bytes
// ---------------------------------------------------------------------------
__global__ void s4_all_kernel(const float* __restrict__ s4B,   // (NL, N, D)
                              const float* __restrict__ s4C,   // (NL, D, N)
                              const float* __restrict__ logdt, // (NL, D)
                              float* __restrict__ K1all) {     // (NL, D, TP)
  extern __shared__ float smem[];
  float* sAdT = smem;                 // [64][66]  (AdT[k][n] = Ad[n][k])
  float* sXC  = smem + 64 * 66;       // [256][66] state xc[d][n]
  float* M1   = sXC;                  // solve scratch (reuses xc region)
  float* M2   = sXC + 64 * 66;
  __shared__ float rowf[64];

  const int layer = blockIdx.x;
  const int tid = threadIdx.x;        // 512 threads = 16 waves

  float dt = expf(logdt[layer * D_]);
  dt = fminf(fmaxf(dt, 1e-4f), 0.1f);

  // M1 = I - dt*A/2 ; M2 = I + dt*A/2  (A = hippo)
  for (int idx = tid; idx < 64 * 64; idx += 512) {
    int r = idx >> 6, c = idx & 63;
    float pr = sqrtf(1.0f + 2.0f * r), pc = sqrtf(1.0f + 2.0f * c);
    float Av = (r > c) ? -pr * pc : ((r == c) ? -((float)r + 0.5f) : pr * pc);
    float dtA = 0.5f * dt * Av;
    float I = (r == c) ? 1.0f : 0.0f;
    M1[r * 66 + c] = I - dtA;
    M2[r * 66 + c] = I + dtA;
  }
  __syncthreads();

  // Gauss-Jordan: solve M1 * Ad = M2  (Ad left in M2)
  {
    int r = tid >> 3;            // 0..63
    int c0 = (tid & 7) * 8;      // 8 columns per thread
    for (int k = 0; k < 64; ++k) {
      float pinv = 1.0f / M1[k * 66 + k];
      __syncthreads();
      if (r == k) {
        for (int j = 0; j < 8; ++j) { M1[k * 66 + c0 + j] *= pinv; M2[k * 66 + c0 + j] *= pinv; }
      }
      __syncthreads();
      if ((tid & 7) == 0) rowf[r] = M1[r * 66 + k];
      __syncthreads();
      if (r != k) {
        float f = rowf[r];
        for (int j = 0; j < 8; ++j) {
          M1[r * 66 + c0 + j] -= f * M1[k * 66 + c0 + j];
          M2[r * 66 + c0 + j] -= f * M2[k * 66 + c0 + j];
        }
      }
      __syncthreads();
    }
  }
  // AdT[k][n] = Ad[n][k]
  for (int idx = tid; idx < 64 * 64; idx += 512) {
    int n = idx >> 6, k = idx & 63;
    sAdT[k * 66 + n] = M2[n * 66 + k];
  }
  __syncthreads();
  // init state xc = Cp  (overwrites solve scratch)
  for (int idx = tid; idx < D_ * NS_; idx += 512) {
    int d = idx >> 6, n = idx & 63;
    sXC[d * 66 + n] = s4C[((size_t)layer * D_ + d) * NS_ + n];
  }
  __syncthreads();

  const int lane = tid & 31, wv = tid >> 5;
  const int m0 = wv * 16, lh = lane >> 4, ln15 = lane & 15;
  const float* BTg = s4B + (size_t)layer * NS_ * D_;   // BT[d][n] = s4B[layer][n][d]

  for (int t = 0; t < T_; ++t) {
    // ks[d] = clip(sum_n xc[d][n] * BT[d][n]) ; k1[d,t] = ks*exp(-0.01 t)
    if (tid < D_) {
      float ssum = 0.0f;
      for (int n = 0; n < NS_; ++n) ssum += sXC[tid * 66 + n] * BTg[n * D_ + tid];
      ssum = fminf(fmaxf(ssum, -10.0f), 10.0f);
      K1all[((size_t)layer * D_ + tid) * TP_ + t] = ssum * expf(-0.01f * (float)t);
    }
    // xn = xc @ AdT  via V_WMMA_F32_16X16X4_F32; wave owns rows m0..m0+15, 4 col tiles
    v8f acc[4];
    for (int p = 0; p < 4; ++p) {
      v8f a = {0.f, 0.f, 0.f, 0.f, 0.f, 0.f, 0.f, 0.f};
      int n0 = p * 16;
      for (int kc = 0; kc < NS_; kc += 4) {
        v2f av = *(const v2f*)&sXC[(m0 + ln15) * 66 + kc + lh * 2];   // A: {K, K+1}
        v2f bv;
        bv.x = sAdT[(kc + lh * 2) * 66 + n0 + ln15];                  // B rows striped by half-lane
        bv.y = sAdT[(kc + 1 + lh * 2) * 66 + n0 + ln15];
        a = __builtin_amdgcn_wmma_f32_16x16x4_f32(false, av, false, bv,
                                                  (short)0, a, false, false);
      }
      acc[p] = a;
    }
    __syncthreads();   // all reads of xc done
    for (int p = 0; p < 4; ++p) {
      int col = p * 16 + ln15;
      for (int v = 0; v < 8; ++v) {
        int row = m0 + v + lh * 8;
        float val = fminf(fmaxf(acc[p][v], -100.0f), 100.0f);
        sXC[row * 66 + col] = val;
      }
    }
    __syncthreads();   // writes visible before next step
  }
}

// ---------------------------------------------------------------------------
// x / (||x||_2 + 1e-8) per (b,t), written transposed to (b*D+d, TP)
// ---------------------------------------------------------------------------
__global__ void normx_kernel(const float* __restrict__ h, float* __restrict__ X) {
  __shared__ float red[256];
  unsigned row = blockIdx.x;
  unsigned b = row / T_, t = row % T_;
  int d = threadIdx.x;
  float v = h[(size_t)row * D_ + d];
  float ss = block_sum_256(v * v, red);
  float xn = v / (sqrtf(ss) + 1e-8f);
  X[((size_t)b * D_ + d) * TP_ + t] = xn;   // scattered write, L2-resident
}

// ---------------------------------------------------------------------------
// causal depthwise conv + gate: y[t] = sum_{tau<=t} k[tau]*x[t-tau] + x[t]*sig(Dp)
// one (b,d) row per block; kernel and zero-padded signal staged in LDS via
// CDNA5 async-to-LDS copies (ASYNCcnt), overlapped with the LDS zero-fill
// ---------------------------------------------------------------------------
__global__ void conv_kernel(const float* __restrict__ X, const float* __restrict__ K1,
                            const float* __restrict__ s4D, float* __restrict__ Y) {
  __shared__ __align__(16) float kk[TP_];
  __shared__ __align__(16) float xs[2 * TP_];
  unsigned bd = blockIdx.x;
  unsigned d = bd & (D_ - 1);
  int tid = threadIdx.x;

  // issue async global->LDS copies first (250 lanes x 16B per row)
  if (tid < T_ / 4) {
    async_load_f4(K1 + (size_t)d * TP_ + tid * 4, &kk[tid * 4]);
    async_load_f4(X + (size_t)bd * TP_ + tid * 4, &xs[TP_ + tid * 4]);
  }
  // zero-fill the pad regions while the async engine streams the payload
  for (int i = tid; i < TP_; i += 256) xs[i] = 0.0f;                    // left zero pad
  for (int i = TP_ + T_ + tid; i < 2 * TP_; i += 256) xs[i] = 0.0f;     // right pad
  if (tid < TP_ - T_) kk[T_ + tid] = 0.0f;                              // kernel tail pad
  async_wait0();
  __syncthreads();

  float a0 = 0.f, a1 = 0.f, a2 = 0.f, a3 = 0.f;
  int t0 = tid;
  for (int tau = 0; tau < T_; ++tau) {
    float kv = kk[tau];
    a0 = fmaf(kv, xs[TP_ + t0 - tau], a0);
    a1 = fmaf(kv, xs[TP_ + t0 + 256 - tau], a1);
    a2 = fmaf(kv, xs[TP_ + t0 + 512 - tau], a2);
    a3 = fmaf(kv, xs[TP_ + t0 + 768 - tau], a3);
  }
  float gate = 1.0f / (1.0f + expf(-s4D[d]));
  Y[(size_t)bd * TP_ + t0]        = a0 + xs[TP_ + t0] * gate;
  Y[(size_t)bd * TP_ + t0 + 256]  = a1 + xs[TP_ + t0 + 256] * gate;
  Y[(size_t)bd * TP_ + t0 + 512]  = a2 + xs[TP_ + t0 + 512] * gate;
  Y[(size_t)bd * TP_ + t0 + 768]  = a3 + xs[TP_ + t0 + 768] * gate;
}

// ---------------------------------------------------------------------------
// h = LN(gelu(y) + 1.1*h)  (exact gelu; in-place per (b,t) row)
// ---------------------------------------------------------------------------
__global__ void fuse_ln_kernel(const float* __restrict__ Y, float* __restrict__ h,
                               const float* __restrict__ g, const float* __restrict__ bta) {
  __shared__ float red[256];
  unsigned row = blockIdx.x;
  unsigned b = row / T_, t = row % T_;
  int d = threadIdx.x;
  float y = Y[((size_t)b * D_ + d) * TP_ + t];
  float ge = 0.5f * y * (1.0f + erff(y * 0.70710678118f));
  float v = ge + 1.1f * h[(size_t)row * D_ + d];
  float mean = block_sum_256(v, red) * (1.0f / D_);
  float dv = v - mean;
  float var = block_sum_256(dv * dv, red) * (1.0f / D_);
  h[(size_t)row * D_ + d] = dv * rsqrtf(var + LN_EPS_) * g[d] + bta[d];
}

// ---------------------------------------------------------------------------
// q = wq @ cls + bq   (identical for all batches)
// ---------------------------------------------------------------------------
__global__ void qproj_kernel(const float* __restrict__ cls, const float* __restrict__ w,
                             const float* __restrict__ bb, float* __restrict__ q) {
  __shared__ float cl[256];
  int tid = threadIdx.x;
  cl[tid] = cls[tid];
  __syncthreads();
  float a = bb[tid];
  for (int k = 0; k < D_; ++k) a = fmaf(cl[k], w[tid * D_ + k], a);
  q[tid] = a;
}

// ---------------------------------------------------------------------------
// single-query attention pooling, one (b, head) per block
// KV layout: KV[r*512 + n], n<256 -> K, n>=256 -> V
// ---------------------------------------------------------------------------
__global__ void attn_pool_kernel(const float* __restrict__ q, const float* __restrict__ KV,
                                 float* __restrict__ pool) {
  __shared__ float sc[TP_];
  __shared__ float red[256];
  __shared__ float qs[HD_];
  int b = blockIdx.x / H_, hh = blockIdx.x % H_;
  int tid = threadIdx.x;
  if (tid < HD_) qs[tid] = q[hh * HD_ + tid];
  __syncthreads();
  const float scale = 0.17677669529f;  // 1/sqrt(32)
  float lmax = -1e30f;
  for (int s = tid; s < S_; s += 256) {
    const float* kr = KV + ((size_t)b * S_ + s) * 512 + hh * HD_;
    float dot = 0.0f;
    for (int j = 0; j < HD_; ++j) dot = fmaf(qs[j], kr[j], dot);
    dot *= scale;
    sc[s] = dot;
    lmax = fmaxf(lmax, dot);
  }
  float mx = block_max_256(lmax, red);
  float lsum = 0.0f;
  float acc[HD_];
  for (int j = 0; j < HD_; ++j) acc[j] = 0.0f;
  for (int s = tid; s < S_; s += 256) {
    float e = expf(sc[s] - mx);
    lsum += e;
    const float* vr = KV + ((size_t)b * S_ + s) * 512 + 256 + hh * HD_;
    for (int j = 0; j < HD_; ++j) acc[j] = fmaf(e, vr[j], acc[j]);
  }
  float sum = block_sum_256(lsum, red);
  float inv = 1.0f / sum;
  for (int j = 0; j < HD_; ++j) {
    float tot = block_sum_256(acc[j], red);
    if (tid == 0) pool[(size_t)b * D_ + hh * HD_ + j] = tot * inv;
  }
}

// ---------------------------------------------------------------------------
// output projection + MLP head, one batch element per block
// ---------------------------------------------------------------------------
__global__ void head_kernel(const float* __restrict__ pool,
                            const float* __restrict__ ow, const float* __restrict__ ob,
                            const float* __restrict__ w1, const float* __restrict__ b1,
                            const float* __restrict__ w2, const float* __restrict__ b2,
                            float* __restrict__ out) {
  __shared__ float p[256];
  __shared__ float p2[256];
  __shared__ float hid[128];
  __shared__ float red[256];
  int b = blockIdx.x, d = threadIdx.x;
  p[d] = pool[(size_t)b * D_ + d];
  __syncthreads();
  float a = ob[d];
  for (int k = 0; k < D_; ++k) a = fmaf(p[k], ow[d * D_ + k], a);
  p2[d] = a;
  __syncthreads();
  if (d < 128) {
    float s = b1[d];
    for (int k = 0; k < D_; ++k) s = fmaf(p2[k], w1[d * D_ + k], s);
    hid[d] = fmaxf(s, 0.0f);
  }
  __syncthreads();
  float z = (d < 128) ? hid[d] * w2[d] : 0.0f;
  float tot = block_sum_256(z, red);
  if (d == 0) out[b] = tot + b2[0];
}

// ---------------------------------------------------------------------------
extern "C" void kernel_launch(void* const* d_in, const int* in_sizes, int n_in,
                              void* d_out, int out_size, void* d_ws, size_t ws_size,
                              hipStream_t stream) {
  (void)in_sizes; (void)n_in; (void)out_size; (void)ws_size;
  const float* x        = (const float*)d_in[0];
  const float* w_in     = (const float*)d_in[1];
  const float* b_in     = (const float*)d_in[2];
  const float* ln_in_g  = (const float*)d_in[3];
  const float* ln_in_b  = (const float*)d_in[4];
  const float* s4_B     = (const float*)d_in[5];
  const float* s4_C     = (const float*)d_in[6];
  const float* s4_logdt = (const float*)d_in[7];
  const float* s4_D     = (const float*)d_in[8];
  const float* ln_g     = (const float*)d_in[9];
  const float* ln_b     = (const float*)d_in[10];
  const float* cls      = (const float*)d_in[11];
  const float* mha_in_w = (const float*)d_in[12];
  const float* mha_in_b = (const float*)d_in[13];
  const float* mha_out_w= (const float*)d_in[14];
  const float* mha_out_b= (const float*)d_in[15];
  const float* head_w1  = (const float*)d_in[16];
  const float* head_b1  = (const float*)d_in[17];
  const float* head_w2  = (const float*)d_in[18];
  const float* head_b2  = (const float*)d_in[19];
  float* out = (float*)d_out;

  float* ws    = (float*)d_ws;
  float* hbuf  = ws + F_H;
  float* xbuf  = ws + F_X;
  float* ybuf  = ws + F_Y;
  float* kvout = ws + F_KV;
  float* k1    = ws + F_K1;
  float* qv    = ws + F_Q;
  float* pool  = ws + F_POOL;
  __bf16* bfr   = (__bf16*)(ws + F_END);
  __bf16* abf   = bfr;                                 // proj A: 32000*160
  __bf16* wbf   = bfr + (size_t)32000 * KP1_;          // proj B^T: 256*160
  __bf16* kvbf  = bfr;                                 // kv A: 32032*256 (reuses proj region)
  __bf16* wkvbf = bfr + (size_t)32032 * 256;           // kv B^T: 512*256

  // input projection (bf16 WMMA GEMM) + LN + positional encoding
  pack_x_kernel<<<20000, 256, 0, stream>>>(x, abf);
  pack_w_kernel<<<160, 256, 0, stream>>>(w_in, wbf);
  gemm_bf16_kernel<<<4000, 256, 0, stream>>>(abf, wbf, b_in, hbuf, 2000, 16, KP1_, D_);
  ln_pos_kernel<<<32000, 256, 0, stream>>>(hbuf, ln_in_g, ln_in_b);

  // S4 kernels for all layers (independent of activations) — f32 WMMA scan
  static const int s4_smem = (64 * 66 + 256 * 66) * 4;   // 84,480 B dynamic LDS
  hipFuncSetAttribute((const void*)s4_all_kernel,
                      hipFuncAttributeMaxDynamicSharedMemorySize, s4_smem);
  s4_all_kernel<<<NL_, 512, s4_smem, stream>>>(s4_B, s4_C, s4_logdt, k1);

  // S4 layers: normalize -> causal conv + gate -> gelu + residual + LN
  for (int i = 0; i < NL_; ++i) {
    normx_kernel<<<32000, 256, 0, stream>>>(hbuf, xbuf);
    conv_kernel<<<B_ * D_, 256, 0, stream>>>(xbuf, k1 + (size_t)i * D_ * TP_,
                                             s4_D + (size_t)i * D_, ybuf);
    fuse_ln_kernel<<<32000, 256, 0, stream>>>(ybuf, hbuf,
                                              ln_g + (size_t)i * D_, ln_b + (size_t)i * D_);
  }

  // attention pooling: K/V projection via bf16 WMMA GEMM, fused softmax pooling
  pack_kv_kernel<<<32032, 256, 0, stream>>>(hbuf, cls, kvbf);
  pack_wkv_kernel<<<512, 256, 0, stream>>>(mha_in_w, wkvbf);
  qproj_kernel<<<1, 256, 0, stream>>>(cls, mha_in_w, mha_in_b, qv);
  gemm_bf16_kernel<<<8008, 256, 0, stream>>>(kvbf, wkvbf, mha_in_b + 256, kvout,
                                             2002, 32, 256, 512);
  attn_pool_kernel<<<B_ * H_, 256, 0, stream>>>(qv, kvout, pool);
  head_kernel<<<B_, 256, 0, stream>>>(pool, mha_out_w, mha_out_b,
                                      head_w1, head_b1, head_w2, head_b2, out);
}
